// SVDQW4A4Linear_53884659696165
// MI455X (gfx1250) — matrix operand order repaired and checked by reading, not compiled, not run on hardware
//
#include <hip/hip_runtime.h>
#include <hip/hip_bf16.h>

#define M_DIM 8192
#define I_DIM 4096
#define O_DIM 4096
#define RANK  32
#define GSZ   64

#define BM 128
#define BN 128
#define BK 64
#define LDA (BK + 8)   // padded f16 stride, 144 B (16B-aligned)
#define LDO (BN + 8)   // padded bf16 stride for output staging, 272 B

typedef __attribute__((ext_vector_type(16))) _Float16 v16h;
typedef __attribute__((ext_vector_type(8)))  _Float16 v8h;
typedef __attribute__((ext_vector_type(8)))  float    v8f;

static __device__ inline v8f vzero8() {
  v8f z;
#pragma unroll
  for (int i = 0; i < 8; ++i) z[i] = 0.0f;
  return z;
}

// 8 packed bf16 (as uint4) -> 8 f16
static __device__ inline v8h bf8_to_h8(uint4 u) {
  unsigned w[4] = {u.x, u.y, u.z, u.w};
  v8h r;
#pragma unroll
  for (int i = 0; i < 4; ++i) {
    union { unsigned u; float f; } lo, hi;
    lo.u = w[i] << 16;
    hi.u = w[i] & 0xffff0000u;
    r[2 * i]     = (_Float16)lo.f;
    r[2 * i + 1] = (_Float16)hi.f;
  }
  return r;
}

static __device__ inline v16h cat16(v8h lo, v8h hi) {
  return __builtin_shufflevector(lo, hi, 0, 1, 2, 3, 4, 5, 6, 7,
                                 8, 9, 10, 11, 12, 13, 14, 15);
}

// CDNA5 async memory->LDS copy, 16 bytes per lane. Tracked by ASYNCcnt.
static __device__ inline void async_ld_b128(unsigned lds_addr, const void* gaddr) {
  asm volatile("global_load_async_to_lds_b128 %0, %1, off"
               :: "v"(lds_addr), "v"(gaddr) : "memory");
}
static __device__ inline void wait_async0() {
  asm volatile("s_wait_asynccnt 0x0" ::: "memory");
}

// ---------------------------------------------------------------------------
// Kernel 1: dequantize int4 codes (stored as int32 0..15) -> f16 W[O][I]
// value = (code - 8) * wscales[k/64][o]
// ---------------------------------------------------------------------------
__global__ __launch_bounds__(256) void dequant_w_kernel(
    const int* __restrict__ qw, const float* __restrict__ wscales,
    _Float16* __restrict__ Wde) {
  size_t base = ((size_t)blockIdx.x * 256 + threadIdx.x) * 8;  // 8 elems/thread
  int o = (int)(base / I_DIM);
  int k = (int)(base % I_DIM);
  float s = wscales[(size_t)(k >> 6) * O_DIM + o];  // GS=64, one group per 8 elems
  int4 q0 = *(const int4*)(qw + base);
  int4 q1 = *(const int4*)(qw + base + 4);
  v8h w;
  w[0] = (_Float16)((float)(q0.x - 8) * s);
  w[1] = (_Float16)((float)(q0.y - 8) * s);
  w[2] = (_Float16)((float)(q0.z - 8) * s);
  w[3] = (_Float16)((float)(q0.w - 8) * s);
  w[4] = (_Float16)((float)(q1.x - 8) * s);
  w[5] = (_Float16)((float)(q1.y - 8) * s);
  w[6] = (_Float16)((float)(q1.z - 8) * s);
  w[7] = (_Float16)((float)(q1.w - 8) * s);
  *(v8h*)(Wde + base) = w;
}

// ---------------------------------------------------------------------------
// Kernel 2: Xs = fp16(x) * fp16(1/smooth)   (bf16 in, f16 out)
// ---------------------------------------------------------------------------
__global__ __launch_bounds__(256) void smooth_x_kernel(
    const unsigned short* __restrict__ x, const float* __restrict__ sm,
    _Float16* __restrict__ Xs) {
  size_t base = ((size_t)blockIdx.x * 256 + threadIdx.x) * 8;
  int k = (int)(base % I_DIM);
  uint4 u = *(const uint4*)(x + base);
  v8h xv = bf8_to_h8(u);
  v8h o;
#pragma unroll
  for (int j = 0; j < 8; ++j) {
    _Float16 rs = (_Float16)(1.0f / sm[k + j]);
    o[j] = xv[j] * rs;  // f16 multiply, as in reference
  }
  *(v8h*)(Xs + base) = o;
}

// ---------------------------------------------------------------------------
// Kernel 3: T[M][RANK] = fp16(x) @ fp16(proj_down)   via WMMA
// block = 128 rows, 8 waves, each wave: 16 rows x 32 cols (2 tiles)
// ---------------------------------------------------------------------------
__global__ __launch_bounds__(256) void lora_down_kernel(
    const unsigned short* __restrict__ x, const float* __restrict__ pd,
    _Float16* __restrict__ T) {
  __shared__ _Float16 PdT[RANK][40];  // [r][kk], chunk of 32 k, padded stride
  const int tid  = threadIdx.x;
  const int lane = tid & 31;
  const int wave = tid >> 5;
  const int ln = lane & 15;
  const int hs = lane >> 4;
  const int m0 = blockIdx.x * 128;

  v8f acc[2];
  acc[0] = vzero8();
  acc[1] = vzero8();

  for (int k0 = 0; k0 < I_DIM; k0 += 32) {
    __syncthreads();
#pragma unroll
    for (int j = 0; j < 4; ++j) {  // 1024 f32 -> f16 LDS transpose
      int idx = tid + j * 256;
      int kk = idx >> 5;
      int r  = idx & 31;
      PdT[r][kk] = (_Float16)pd[(size_t)(k0 + kk) * RANK + r];
    }
    __syncthreads();

    // A fragment straight from global x (bf16), converted to f16
    const unsigned short* xp =
        x + (size_t)(m0 + wave * 16 + ln) * I_DIM + k0;
    uint4 ulo = *(const uint4*)(xp + hs * 8);
    uint4 uhi = *(const uint4*)(xp + 16 + hs * 8);
    v16h a = cat16(bf8_to_h8(ulo), bf8_to_h8(uhi));

#pragma unroll
    for (int ni = 0; ni < 2; ++ni) {
      v8h blo = *(const v8h*)&PdT[ni * 16 + ln][hs * 8];
      v8h bhi = *(const v8h*)&PdT[ni * 16 + ln][16 + hs * 8];
      v16h b = cat16(blo, bhi);
      acc[ni] = __builtin_amdgcn_wmma_f32_16x16x32_f16(
          false, a, false, b, (short)0, acc[ni], false, false);
    }
  }

#pragma unroll
  for (int ni = 0; ni < 2; ++ni)
#pragma unroll
    for (int j = 0; j < 8; ++j) {
      int row = m0 + wave * 16 + j + hs * 8;
      int col = ni * 16 + ln;
      T[(size_t)row * RANK + col] = (_Float16)acc[ni][j];
    }
}

// ---------------------------------------------------------------------------
// Kernel 4: main GEMM  out = Xs @ Wde^T + T @ (wt*proj_up)^T + bias  -> bf16
// 128x128 tile / block, 8 waves; wave = 32x64 region = 2x4 WMMA tiles.
// Double-buffered async (memory->LDS) staging; LDS-transposed b128 stores.
// ---------------------------------------------------------------------------
__global__ __launch_bounds__(256) void wmma_main_kernel(
    const _Float16* __restrict__ Xs, const _Float16* __restrict__ Wde,
    const _Float16* __restrict__ T, const float* __restrict__ proj_up,
    const float* __restrict__ wtscale, const float* __restrict__ bias,
    __hip_bfloat16* __restrict__ out) {
  // 2 x (A + B) double-buffered f16 tiles = 73,728 B; output staging overlays.
  __shared__ __align__(16) unsigned char smem[4 * BM * LDA * 2];
  typedef _Float16 tileT[BM][LDA];
  tileT* As = (tileT*)smem;                      // As[0..1][BM][LDA]
  tileT* Bs = (tileT*)(smem + 2 * BM * LDA * 2); // Bs[0..1][BM][LDA]
  typedef __hip_bfloat16 orowT[LDO];
  orowT* Ot = (orowT*)smem;                      // 128 x 136 bf16 = 34,816 B

  const int tid  = threadIdx.x;
  const int lane = tid & 31;
  const int wave = tid >> 5;
  const int wm = wave >> 1;   // 0..3 : 32-row group
  const int wn = wave & 1;    // 0..1 : 64-col group
  const int ln = lane & 15;
  const int hs = lane >> 4;
  const int m0 = blockIdx.y * BM;
  const int n0 = blockIdx.x * BN;

  v8f acc[2][4];
#pragma unroll
  for (int mi = 0; mi < 2; ++mi)
#pragma unroll
    for (int ni = 0; ni < 4; ++ni) acc[mi][ni] = vzero8();

  // async stage of one 128x64 A chunk + one 128x64 B chunk into buffer `buf`
  auto stage = [&](int kt, int buf) {
#pragma unroll
    for (int j = 0; j < 4; ++j) {
      int idx = tid * 8 + j * 2048;  // 8192 f16 per matrix, 32/thread
      int r = idx >> 6;
      int c = idx & 63;
      async_ld_b128((unsigned)(size_t)&As[buf][r][c],
                    (const void*)(Xs + (size_t)(m0 + r) * I_DIM + kt + c));
      async_ld_b128((unsigned)(size_t)&Bs[buf][r][c],
                    (const void*)(Wde + (size_t)(n0 + r) * I_DIM + kt + c));
    }
  };

  stage(0, 0);
  for (int kt = 0; kt < I_DIM; kt += BK) {
    const int buf = (kt >> 6) & 1;
    wait_async0();    // my chunk-kt loads have landed in LDS
    __syncthreads();  // everyone's chunk-kt loads have landed
    if (kt + BK < I_DIM) stage(kt + BK, buf ^ 1);  // prefetch under compute

#pragma unroll
    for (int ks = 0; ks < BK; ks += 32) {
      v16h a[2], b[4];
#pragma unroll
      for (int mi = 0; mi < 2; ++mi) {
        int row = wm * 32 + mi * 16 + ln;
        v8h alo = *(const v8h*)&As[buf][row][ks + hs * 8];
        v8h ahi = *(const v8h*)&As[buf][row][ks + 16 + hs * 8];
        a[mi] = cat16(alo, ahi);
      }
#pragma unroll
      for (int ni = 0; ni < 4; ++ni) {
        int col = wn * 64 + ni * 16 + ln;
        v8h blo = *(const v8h*)&Bs[buf][col][ks + hs * 8];
        v8h bhi = *(const v8h*)&Bs[buf][col][ks + 16 + hs * 8];
        b[ni] = cat16(blo, bhi);
      }
#pragma unroll
      for (int mi = 0; mi < 2; ++mi)
#pragma unroll
        for (int ni = 0; ni < 4; ++ni)
          acc[mi][ni] = __builtin_amdgcn_wmma_f32_16x16x32_f16(
              false, a[mi], false, b[ni], (short)0, acc[mi][ni], false, false);
    }
  }

  // -------- Epilogue: LoRA up-projection (one WMMA per tile, K=RANK=32) ----
  float wt = wtscale[0];
  v16h at[2];
#pragma unroll
  for (int mi = 0; mi < 2; ++mi) {
    int row = m0 + wm * 32 + mi * 16 + ln;
    v8h lo = *(const v8h*)(T + (size_t)row * RANK + hs * 8);
    v8h hi = *(const v8h*)(T + (size_t)row * RANK + 16 + hs * 8);
    at[mi] = cat16(lo, hi);
  }
#pragma unroll
  for (int ni = 0; ni < 4; ++ni) {
    int col = n0 + wn * 64 + ni * 16 + ln;
    const float* pu = proj_up + (size_t)col * RANK + hs * 8;
    v16h bt;
#pragma unroll
    for (int j = 0; j < 16; ++j) bt[j] = (_Float16)(pu[(j & 7) + (j >> 3) * 16] * wt);
#pragma unroll
    for (int mi = 0; mi < 2; ++mi)
      acc[mi][ni] = __builtin_amdgcn_wmma_f32_16x16x32_f16(
          false, at[mi], false, bt, (short)0, acc[mi][ni], false, false);
    float bv = bias[col];
#pragma unroll
    for (int mi = 0; mi < 2; ++mi)
#pragma unroll
      for (int j = 0; j < 8; ++j) acc[mi][ni][j] += bv;
  }

  // -------- Store: transpose through LDS, then coalesced b128 stores -------
  __syncthreads();  // done reading As/Bs; safe to overlay Ot
#pragma unroll
  for (int mi = 0; mi < 2; ++mi)
#pragma unroll
    for (int ni = 0; ni < 4; ++ni) {
      int cl = wn * 64 + ni * 16 + ln;
#pragma unroll
      for (int j = 0; j < 8; ++j) {
        int rl = wm * 32 + mi * 16 + j + hs * 8;
        Ot[rl][cl] = __float2bfloat16(acc[mi][ni][j]);
      }
    }
  __syncthreads();
#pragma unroll
  for (int j = 0; j < 8; ++j) {  // 16384 bf16 / 256 thr = 64 each = 8 x b128
    int idx = tid * 8 + j * 2048;
    int r = idx >> 7;
    int c = idx & 127;
    uint4 v = *(const uint4*)&Ot[r][c];
    *(uint4*)(out + (size_t)(m0 + r) * O_DIM + n0 + c) = v;
  }
}

// ---------------------------------------------------------------------------
extern "C" void kernel_launch(void* const* d_in, const int* in_sizes, int n_in,
                              void* d_out, int out_size, void* d_ws,
                              size_t ws_size, hipStream_t stream) {
  const unsigned short* x  = (const unsigned short*)d_in[0];  // bf16 [M,I]
  const int*   qweight     = (const int*)d_in[1];             // [O,I]
  const float* wscales     = (const float*)d_in[2];           // [I/GS,O]
  const float* smooth      = (const float*)d_in[3];           // [I]
  const float* proj_down   = (const float*)d_in[4];           // [I,R]
  const float* proj_up     = (const float*)d_in[5];           // [O,R]
  const float* wtscale     = (const float*)d_in[6];           // [1]
  const float* bias        = (const float*)d_in[7];           // [O]
  __hip_bfloat16* out      = (__hip_bfloat16*)d_out;          // [M,O]

  char* ws = (char*)d_ws;
  _Float16* Wde = (_Float16*)ws;                                   // 32 MB
  _Float16* Xs  = (_Float16*)(ws + (size_t)O_DIM * I_DIM * 2);     // 64 MB
  _Float16* T   = (_Float16*)(ws + (size_t)O_DIM * I_DIM * 2 +
                              (size_t)M_DIM * I_DIM * 2);          // 0.5 MB

  dequant_w_kernel<<<dim3((O_DIM * (size_t)I_DIM) / 2048), 256, 0, stream>>>(
      qweight, wscales, Wde);
  smooth_x_kernel<<<dim3((M_DIM * (size_t)I_DIM) / 2048), 256, 0, stream>>>(
      x, smooth, Xs);
  lora_down_kernel<<<dim3(M_DIM / 128), 256, 0, stream>>>(x, proj_down, T);
  wmma_main_kernel<<<dim3(O_DIM / BN, M_DIM / BM), 256, 0, stream>>>(
      Xs, Wde, T, proj_up, wtscale, bias, out);
}
